// Head_2980707303948
// MI455X (gfx1250) — matrix operand order repaired and checked by reading, not compiled, not run on hardware
//
#include <hip/hip_runtime.h>

#define EMBED 1024
#define HEADD 64
#define TSEQ  4096
#define NROWS 16384  // B*T

typedef _Float16 half4v  __attribute__((ext_vector_type(4)));
typedef _Float16 half8v  __attribute__((ext_vector_type(8)));
typedef _Float16 half16v __attribute__((ext_vector_type(16)));
typedef float    float8v __attribute__((ext_vector_type(8)));

__device__ __forceinline__ half16v cmb(half8v lo, half8v hi) {
  return __builtin_shufflevector(lo, hi, 0,1,2,3,4,5,6,7,8,9,10,11,12,13,14,15);
}
__device__ __forceinline__ half8v ld8(const _Float16* p) {
  return *(const half8v*)p;
}
__device__ __forceinline__ float8v wmma16(half16v a, half16v b, float8v c) {
  return __builtin_amdgcn_wmma_f32_16x16x32_f16(false, a, false, b, (short)0, c,
                                                false, false);
}
// 16 contiguous f32 -> f16 B-fragment slice
__device__ __forceinline__ half16v ldw16(const float* p) {
  float4 a = *(const float4*)(p);
  float4 b = *(const float4*)(p + 4);
  float4 c = *(const float4*)(p + 8);
  float4 d = *(const float4*)(p + 12);
  half16v h;
  h[0]=(_Float16)a.x;  h[1]=(_Float16)a.y;  h[2]=(_Float16)a.z;  h[3]=(_Float16)a.w;
  h[4]=(_Float16)b.x;  h[5]=(_Float16)b.y;  h[6]=(_Float16)b.z;  h[7]=(_Float16)b.w;
  h[8]=(_Float16)c.x;  h[9]=(_Float16)c.y;  h[10]=(_Float16)c.z; h[11]=(_Float16)c.w;
  h[12]=(_Float16)d.x; h[13]=(_Float16)d.y; h[14]=(_Float16)d.z; h[15]=(_Float16)d.w;
  return h;
}
// low 32 bits of a flat shared pointer == LDS byte address
__device__ __forceinline__ unsigned ldsoff(const void* p) {
  return (unsigned)(unsigned long)p;
}

// ---------------------------------------------------------------------------
// Kernel 1: q/k/v = x @ W^T, f16 outputs. One block = 16 rows of x, 4 waves,
// wave w owns output columns [16w, 16w+16) for all three weight matrices.
// ---------------------------------------------------------------------------
__global__ __launch_bounds__(128) void qkv_proj_kernel(
    const float* __restrict__ x,
    const float* __restrict__ Wq,
    const float* __restrict__ Wk,
    const float* __restrict__ Wv,
    _Float16* __restrict__ qb,   // [NROWS][64]
    _Float16* __restrict__ kb,   // [NROWS][64]
    _Float16* __restrict__ vtb)  // [B][64][T]  (transposed V)
{
  __shared__ __align__(16) _Float16 xs[16 * EMBED];  // 32KB

  const int tid    = threadIdx.x;
  const int wave   = tid >> 5;
  const int lane   = tid & 31;
  const int lane16 = lane & 15;
  const int hlf    = lane >> 4;
  const long m0    = (long)blockIdx.x * 16;

  // stage 16x1024 f32 tile -> f16 in LDS
  {
    const float4* src = (const float4*)(x + m0 * EMBED);
    half4v* dst = (half4v*)xs;
#pragma unroll
    for (int i = 0; i < 32; ++i) {  // 4096 float4 / 128 threads
      int idx = tid + i * 128;
      float4 f = src[idx];
      half4v h;
      h[0] = (_Float16)f.x; h[1] = (_Float16)f.y;
      h[2] = (_Float16)f.z; h[3] = (_Float16)f.w;
      dst[idx] = h;
    }
  }
  __syncthreads();

  const int hcol = wave * 16 + lane16;  // output column 0..63
  const float* wq = Wq + (long)hcol * EMBED + hlf * 16;
  const float* wk = Wk + (long)hcol * EMBED + hlf * 16;
  const float* wv = Wv + (long)hcol * EMBED + hlf * 16;

  float8v accq = {0,0,0,0,0,0,0,0};
  float8v acck = accq, accv = accq;

  for (int kk = 0; kk < EMBED / 32; ++kk) {
    // A fragment: row = lane16, K elems {16h..16h+15}
    const _Float16* ar = xs + lane16 * EMBED + kk * 32 + hlf * 16;
    half16v afrag = cmb(ld8(ar), ld8(ar + 8));

    half16v fq = ldw16(wq + kk * 32);
    half16v fk = ldw16(wk + kk * 32);
    half16v fv = ldw16(wv + kk * 32);

    accq = wmma16(afrag, fq, accq);
    acck = wmma16(afrag, fk, acck);
    accv = wmma16(afrag, fv, accv);
  }

  // q,k row-major [row][64]
#pragma unroll
  for (int r = 0; r < 8; ++r) {
    long grow = m0 + r + 8 * hlf;  // C/D layout: M = reg + 8*half
    qb[grow * HEADD + hcol] = (_Float16)accq[r];
    kb[grow * HEADD + hcol] = (_Float16)acck[r];
  }
  // v transposed [b][h][t]: 8 consecutive t per lane -> one 16B store
  {
    const int bidx = (int)(m0 >> 12);               // T = 4096
    const int t0   = (int)(m0 & (TSEQ - 1)) + 8 * hlf;
    half8v vh;
#pragma unroll
    for (int r = 0; r < 8; ++r) vh[r] = (_Float16)accv[r];
    *(half8v*)(vtb + ((long)(bidx * HEADD + hcol)) * TSEQ + t0) = vh;
  }
}

// ---------------------------------------------------------------------------
// Kernel 2: flash attention. 8 waves/block, each wave = 16 query rows.
// Double-buffered K/V^T panels staged via global_load_async_to_lds_b128
// (ASYNCcnt), next panel prefetched while the current one is computed.
// ---------------------------------------------------------------------------
__global__ __launch_bounds__(256) void flash_attn_kernel(
    const _Float16* __restrict__ qb,
    const _Float16* __restrict__ kb,
    const _Float16* __restrict__ vtb,
    float* __restrict__ out)
{
  __shared__ __align__(16) _Float16 ksh[2][32 * HEADD];  // [key][dim]  2x4KB
  __shared__ __align__(16) _Float16 vsh[2][HEADD * 32];  // [dim][key]  2x4KB
  __shared__ __align__(16) _Float16 psh[8 * 16 * 32];    // per-wave P  8KB

  const int tid    = threadIdx.x;
  const int wave   = tid >> 5;
  const int lane   = tid & 31;
  const int lane16 = lane & 15;
  const int hlf    = lane >> 4;
  const int vd     = tid >> 2;   // staging: dim row of V^T
  const int vg     = tid & 3;    // staging: 8-key group

  const int  tile  = blockIdx.x * 8 + wave;  // 0..1023 (16-row q tiles)
  const int  bb    = blockIdx.x >> 5;        // 32 blocks per batch
  const long qrow0 = (long)tile * 16;        // global row of first query

  const _Float16* kbase = kb  + (long)bb * TSEQ * HEADD;
  const _Float16* vbase = vtb + (long)bb * HEADD * TSEQ;

  // Q A-fragments for dims 0..31 and 32..63, pre-scaled by 1/8 (exact in f16)
  const _Float16* qp = qb + (qrow0 + lane16) * HEADD + hlf * 16;
  half16v aq0 = cmb(ld8(qp),      ld8(qp + 8));
  half16v aq1 = cmb(ld8(qp + 32), ld8(qp + 40));
  aq0 *= (_Float16)0.125f;
  aq1 *= (_Float16)0.125f;

  float8v o0 = {0,0,0,0,0,0,0,0};
  float8v o1 = o0, o2 = o0, o3 = o0;
  float mrow[8], lrow[8];
#pragma unroll
  for (int r = 0; r < 8; ++r) { mrow[r] = -1e30f; lrow[r] = 0.0f; }

  _Float16* pw = psh + wave * 512;

  // async stage of one 32-key panel into buffer `buf`:
  // each of the 256 threads moves one 16B chunk of K and one of V^T.
  // tid*8 == vd*32 + vg*8, so the same index covers both layouts.
#define STAGE_PANEL(buf, kb0)                                                  \
  {                                                                            \
    unsigned kl = ldsoff(&ksh[buf][tid * 8]);                                  \
    unsigned long kg =                                                         \
        (unsigned long)(kbase + (long)(kb0) * HEADD + tid * 8);                \
    asm volatile("global_load_async_to_lds_b128 %0, %1, off" ::"v"(kl),        \
                 "v"(kg)                                                       \
                 : "memory");                                                  \
    unsigned vl = ldsoff(&vsh[buf][tid * 8]);                                  \
    unsigned long vg64 =                                                       \
        (unsigned long)(vbase + (long)vd * TSEQ + (kb0) + vg * 8);             \
    asm volatile("global_load_async_to_lds_b128 %0, %1, off" ::"v"(vl),        \
                 "v"(vg64)                                                     \
                 : "memory");                                                  \
  }

  STAGE_PANEL(0, 0)
  asm volatile("s_wait_asynccnt 0" ::: "memory");
  __syncthreads();

  for (int j = 0; j < TSEQ / 32; ++j) {
    const int cur = j & 1;
    if (j + 1 < TSEQ / 32) STAGE_PANEL(cur ^ 1, (j + 1) * 32)

    // ---- S = (Q/8) K^T : two 16-key n-tiles ----
    float8v s0 = {0,0,0,0,0,0,0,0};
    float8v s1 = s0;
    {
      const _Float16* k0 = ksh[cur] + lane16 * HEADD + hlf * 16;        // keys 0-15
      const _Float16* k1 = ksh[cur] + (16 + lane16) * HEADD + hlf * 16; // keys 16-31
      half16v b00 = cmb(ld8(k0),      ld8(k0 + 8));   // dims 0..31
      half16v b01 = cmb(ld8(k0 + 32), ld8(k0 + 40));  // dims 32..63
      half16v b10 = cmb(ld8(k1),      ld8(k1 + 8));
      half16v b11 = cmb(ld8(k1 + 32), ld8(k1 + 40));
      s0 = wmma16(aq0, b00, s0);
      s0 = wmma16(aq1, b01, s0);
      s1 = wmma16(aq0, b10, s1);
      s1 = wmma16(aq1, b11, s1);
    }

    // ---- online softmax (row = reg + 8*half; 16 cols across a lane-half) ----
    float alpha[8];
#pragma unroll
    for (int r = 0; r < 8; ++r) {
      float v0 = s0[r], v1 = s1[r];
      float mx = fmaxf(v0, v1);
#pragma unroll
      for (int sh = 1; sh <= 8; sh <<= 1)
        mx = fmaxf(mx, __shfl_xor(mx, sh, 32));
      float mnew = fmaxf(mrow[r], mx);
      float a  = __expf(mrow[r] - mnew);
      float p0 = __expf(v0 - mnew);
      float p1 = __expf(v1 - mnew);
      float sum = p0 + p1;
#pragma unroll
      for (int sh = 1; sh <= 8; sh <<= 1)
        sum += __shfl_xor(sum, sh, 32);
      lrow[r] = lrow[r] * a + sum;
      mrow[r] = mnew;
      alpha[r] = a;
      int prow = r + 8 * hlf;                 // C-layout row -> P row
      pw[prow * 32 + lane16]      = (_Float16)p0;
      pw[prow * 32 + 16 + lane16] = (_Float16)p1;
    }
#pragma unroll
    for (int r = 0; r < 8; ++r) {
      o0[r] *= alpha[r]; o1[r] *= alpha[r];
      o2[r] *= alpha[r]; o3[r] *= alpha[r];
    }

    // P f32(C-layout) -> f16 A-fragment via per-wave LDS bounce
    asm volatile("s_wait_dscnt 0" ::: "memory");
    const _Float16* pp = pw + lane16 * 32 + hlf * 16;
    half16v ap = cmb(ld8(pp), ld8(pp + 8));

    // ---- O += P @ V : four 16-dim n-tiles ----
    {
      const _Float16* v0p = vsh[cur] + lane16 * 32 + hlf * 16;
      o0 = wmma16(ap, cmb(ld8(v0p), ld8(v0p + 8)), o0);
      const _Float16* v1p = vsh[cur] + (16 + lane16) * 32 + hlf * 16;
      o1 = wmma16(ap, cmb(ld8(v1p), ld8(v1p + 8)), o1);
      const _Float16* v2p = vsh[cur] + (32 + lane16) * 32 + hlf * 16;
      o2 = wmma16(ap, cmb(ld8(v2p), ld8(v2p + 8)), o2);
      const _Float16* v3p = vsh[cur] + (48 + lane16) * 32 + hlf * 16;
      o3 = wmma16(ap, cmb(ld8(v3p), ld8(v3p + 8)), o3);
    }

    // own async loads landed; then block-wide visibility for next iteration
    asm volatile("s_wait_asynccnt 0" ::: "memory");
    __syncthreads();
  }

  // epilogue: out[b,t,h] = O / l  (fp32)
#pragma unroll
  for (int r = 0; r < 8; ++r) {
    float inv = 1.0f / lrow[r];
    long trow = (qrow0 + r + 8 * hlf) * HEADD;
    out[trow + lane16]      = o0[r] * inv;
    out[trow + 16 + lane16] = o1[r] * inv;
    out[trow + 32 + lane16] = o2[r] * inv;
    out[trow + 48 + lane16] = o3[r] * inv;
  }
}

extern "C" void kernel_launch(void* const* d_in, const int* in_sizes, int n_in,
                              void* d_out, int out_size, void* d_ws, size_t ws_size,
                              hipStream_t stream) {
  const float* x  = (const float*)d_in[0];
  const float* Wk = (const float*)d_in[1];
  const float* Wq = (const float*)d_in[2];
  const float* Wv = (const float*)d_in[3];
  float* out = (float*)d_out;

  _Float16* qbuf = (_Float16*)d_ws;                 // 2MB
  _Float16* kbuf = qbuf + (long)NROWS * HEADD;      // 2MB
  _Float16* vtb  = kbuf + (long)NROWS * HEADD;      // 2MB

  qkv_proj_kernel<<<NROWS / 16, 128, 0, stream>>>(x, Wq, Wk, Wv,
                                                  qbuf, kbuf, vtb);
  flash_attn_kernel<<<NROWS / 16 / 8, 256, 0, stream>>>(qbuf, kbuf, vtb, out);
}
